// AQLProposalNet_13443247637114
// MI455X (gfx1250) — matrix operand order repaired and checked by reading, not compiled, not run on hardware
//
#include <hip/hip_runtime.h>
#include <stdint.h>

// Problem constants (from the reference)
#define B_ROWS 1024
#define DIM    64
#define NACT   100000
#define KSEL   100

// ---- WMMA vector types (gfx1250, wave32) ----
typedef __attribute__((ext_vector_type(16))) __bf16 v16bf;
typedef __attribute__((ext_vector_type(8)))  float  v8f;
// ---- SGPR tuples for the Tensor DMA descriptor (D#) ----
typedef __attribute__((ext_vector_type(4))) uint32_t u32x4;
typedef __attribute__((ext_vector_type(8))) uint32_t u32x8;

// ---- 32-bit counter-based RNG -> Gumbel (deterministic, recomputable) ----
__device__ __forceinline__ uint32_t hash32(uint32_t x) {
    x ^= x >> 17;  x *= 0xED5AD4BBu;
    x ^= x >> 11;  x *= 0xAC4C1B51u;
    x ^= x >> 15;  x *= 0x31848BABu;
    x ^= x >> 14;
    return x;
}

__device__ __forceinline__ float gumbelf(uint32_t row, uint32_t col, uint32_t salt) {
    // row*NACT+col < 1024*100000 < 2^32 : fits one 32-bit counter
    uint32_t h = hash32((row * (uint32_t)NACT + col) ^ (0x9E3779B9u * (salt + 1u)));
    float u = ((float)(h >> 8) + 0.5f) * (1.0f / 16777216.0f);   // (0,1)
    return -__logf(-__logf(u));
}

// Monotone float -> uint32 key (larger float => larger key)
__device__ __forceinline__ uint32_t fkey(float f) {
    uint32_t u = __float_as_uint(f);
    return (u & 0x80000000u) ? ~u : (u | 0x80000000u);
}

// Generic pointer -> LDS byte offset. Per the CDNA5 aperture layout the low
// 32 bits of a flat shared-aperture address ARE the LDS offset; the ptrtoint
// also *captures* the LDS object, so the optimizer must assume the inline-asm
// "memory" clobber can write it (prevents undef-folding of later LDS reads).
__device__ __forceinline__ uint32_t lds_off_of(const void* p) {
    return (uint32_t)(uintptr_t)p;
}

// =====================================================================
// Kernel 1: scores[b,i] = (s @ embd^T)[b,i] + gumbel_exploit(b,i)
// Block = 16 rows x 64 cols, 4 waves; each wave one 16x16 tile via
// 2x v_wmma_f32_16x16x32_bf16 (K = 64).
//   embd tile (64x64 f32) : TDM tensor_load_to_lds (pad -> pitch 68)
//   s    tile (16x64 f32) : global_load_async_to_lds_b128
// =====================================================================
#define TILE_N    64
#define PITCH     68                        // floats; 64 + 4 pad (TDM pad_amount)
#define S_OFF     (TILE_N * PITCH)          // float offset of s tile in LDS
#define LDS_FLTS  ((TILE_N + 16) * PITCH)   // 5440 floats = 21,760 B

__global__ __launch_bounds__(128)
void score_gemm_kernel(const float* __restrict__ s,
                       const float* __restrict__ embd,
                       float* __restrict__ scores)
{
    __shared__ float smem[LDS_FLTS];

    const int tid     = threadIdx.x;
    const int wave    = tid >> 5;
    const int lane    = tid & 31;
    const int colBase = blockIdx.x * TILE_N;
    const int rowBase = blockIdx.y * 16;

    // ---- TDM: one instruction stages the whole 64x64 embd tile ----------
    if (wave == 0) {
        const uint64_t gaddr = (uint64_t)(uintptr_t)(embd + (size_t)colBase * DIM);
        const uint32_t ldsA  = lds_off_of(&smem[0]);  // captures smem
        u32x4 g0;
        g0[0] = 1u;                                   // count=1, no gather
        g0[1] = ldsA;                                 // lds_addr (bytes)
        g0[2] = (uint32_t)gaddr;                      // global_addr[31:0]
        g0[3] = (uint32_t)((gaddr >> 32) & 0x1FFFFFFu) | (2u << 30); // addr[56:32], type=2
        u32x8 g1;
        g1[0] = (2u << 16)          // data_size = 4 bytes
              | (1u << 20)          // pad_enable
              | (5u << 22)          // pad_interval: 64 DWORDs
              | (3u << 25);         // pad_amount  : 4 DWORDs  -> pitch 68 floats
        g1[1] = (uint32_t)DIM << 16;                  // tensor_dim0[15:0] = 64
        g1[2] = ((uint32_t)NACT & 0xFFFFu) << 16;     // dim0[31:16]=0 | dim1[15:0]
        g1[3] = ((uint32_t)NACT >> 16)                // tensor_dim1[31:16]
              | ((uint32_t)DIM << 16);                // tile_dim0 = 64
        g1[4] = (uint32_t)TILE_N;                     // tile_dim1 = 64, tile_dim2 = 0
        g1[5] = (uint32_t)DIM;                        // tensor_dim0_stride = 64
        g1[6] = 0u;                                   // stride hi, dim1_stride lo
        g1[7] = 0u;                                   // dim1_stride hi (unused, 2D)
        asm volatile("tensor_load_to_lds %0, %1" :: "s"(g0), "s"(g1) : "memory");
    }

    // ---- async-LDS: stage the 16x64 s tile (256 b128 moves) -------------
    for (int q = tid; q < 16 * (DIM / 4); q += 128) {
        int r   = q >> 4;
        int seg = q & 15;
        unsigned long long ga =
            (unsigned long long)(uintptr_t)(s + (size_t)(rowBase + r) * DIM + seg * 4);
        uint32_t ldsd = lds_off_of(&smem[S_OFF + r * PITCH + seg * 4]); // captures smem
        asm volatile("global_load_async_to_lds_b128 %0, %1, off"
                     :: "v"(ldsd), "v"(ga) : "memory");
    }
    asm volatile("s_wait_asynccnt 0" ::: "memory");
    if (wave == 0) __builtin_amdgcn_s_wait_tensorcnt(0);
    __syncthreads();

    // ---- A fragments from LDS s tile (f32 -> bf16), 16x32 ISA layout ----
    const int mA = lane & 15;
    const int hA = lane >> 4;
    const float* srow = &smem[S_OFF + mA * PITCH];
    v16bf afrag[2];
#pragma unroll
    for (int kk = 0; kk < 2; ++kk) {
        const int kbase = kk * 32;
#pragma unroll
        for (int j = 0; j < 8; ++j) {
            int k0 = kbase + ((j < 4) ? (hA * 8 + 2 * j)
                                      : (16 + hA * 8 + 2 * (j - 4)));
            afrag[kk][2 * j]     = (__bf16)srow[k0];
            afrag[kk][2 * j + 1] = (__bf16)srow[k0 + 1];
        }
    }

    // ---- B fragments from LDS embd tile (rows = output cols), 32x16 -----
    const int nB = lane & 15;
    const int hB = lane >> 4;
    const float* brow = &smem[(wave * 16 + nB) * PITCH];
    v8f acc = {};
#pragma unroll
    for (int kk = 0; kk < 2; ++kk) {
        const int kbase = kk * 32;
        v16bf bfrag;
#pragma unroll
        for (int j = 0; j < 8; ++j) {
            int k0 = kbase + ((j < 4) ? (hB * 8 + 2 * j)
                                      : (16 + hB * 8 + 2 * (j - 4)));
            bfrag[2 * j]     = (__bf16)brow[k0];
            bfrag[2 * j + 1] = (__bf16)brow[k0 + 1];
        }
        acc = __builtin_amdgcn_wmma_f32_16x16x32_bf16(
                  false, afrag[kk], false, bfrag,
                  (short)0, acc, false, false);
    }

    // ---- D layout: n = lane&15 ; VGPR j -> m = j + 8*(lane>=16) ---------
    const int nOut  = lane & 15;
    const int mHalf = (lane >> 4) * 8;
    const int col   = colBase + wave * 16 + nOut;
    if (col < NACT) {
#pragma unroll
        for (int j = 0; j < 8; ++j) {
            int row = rowBase + mHalf + j;
            scores[(size_t)row * NACT + col] = acc[j] + gumbelf(row, col, 0u);
        }
    }
}

// =====================================================================
// Kernel 2: exact per-row 100th-largest via 3-pass radix select
// (12 + 12 + 8 bits) with LDS histograms (ds_add atomics).
// blockIdx.y == 0 : exploit stream (scores from ws)
// blockIdx.y == 1 : explore stream (pure gumbel, recomputed, zero memory)
// =====================================================================
__global__ __launch_bounds__(256)
void select_threshold_kernel(const float* __restrict__ scores,
                             uint32_t* __restrict__ thr)
{
    __shared__ uint32_t hist[4096];          // 16 KB
    __shared__ uint32_t sh_b, sh_k;

    const int row   = blockIdx.x;
    const int which = blockIdx.y;
    const int tid   = threadIdx.x;

    const int shifts[3] = {20, 8, 0};
    const int bitsc[3]  = {12, 12, 8};

    uint32_t prefix = 0, prefmask = 0, kneed = KSEL;

    for (int p = 0; p < 3; ++p) {
        const int      sh = shifts[p];
        const uint32_t nb = 1u << bitsc[p];

        for (uint32_t j = tid; j < nb; j += 256) hist[j] = 0;
        __syncthreads();

        for (int i = tid; i < NACT; i += 256) {
            float v;
            if (which == 0) {
                if (i + 2048 < NACT)
                    __builtin_prefetch(&scores[(size_t)row * NACT + i + 2048], 0, 0);
                v = scores[(size_t)row * NACT + i];
            } else {
                v = gumbelf(row, i, 1u);
            }
            uint32_t k = fkey(v);
            if ((k & prefmask) == prefix)
                atomicAdd(&hist[(k >> sh) & (nb - 1u)], 1u);
        }
        __syncthreads();

        if (tid == 0) {
            uint32_t cum = 0;
            int b = (int)nb - 1;
            for (;; --b) {
                cum += hist[b];
                if (cum >= kneed || b == 0) break;
            }
            sh_b = (uint32_t)b;
            sh_k = (cum >= kneed) ? (kneed - (cum - hist[b])) : 1u;
        }
        __syncthreads();

        prefix   |= sh_b << sh;
        prefmask |= (nb - 1u) << sh;
        kneed     = sh_k;
        __syncthreads();
    }

    if (tid == 0) thr[which * B_ROWS + row] = prefix;   // exact k-th largest key
}

// =====================================================================
// Kernel 3: mask[b,i] = (key(scores) >= t1[b]) || (key(gumbel2) >= t2[b])
// float4-vectorized read/write; explore gumbel recomputed on the fly.
// =====================================================================
__global__ __launch_bounds__(256)
void mask_kernel(const float* __restrict__ scores,
                 const uint32_t* __restrict__ thr,
                 float* __restrict__ mask)
{
    const int row = blockIdx.y;
    const uint32_t t1 = thr[row];
    const uint32_t t2 = thr[B_ROWS + row];
    const int i0 = (blockIdx.x * 256 + threadIdx.x) * 4;

    if (i0 + 3 < NACT) {
        const float4 sc = *(const float4*)(scores + (size_t)row * NACT + i0);
        float4 out;
        out.x = ((fkey(sc.x) >= t1) || (fkey(gumbelf(row, i0 + 0, 1u)) >= t2)) ? 1.0f : 0.0f;
        out.y = ((fkey(sc.y) >= t1) || (fkey(gumbelf(row, i0 + 1, 1u)) >= t2)) ? 1.0f : 0.0f;
        out.z = ((fkey(sc.z) >= t1) || (fkey(gumbelf(row, i0 + 2, 1u)) >= t2)) ? 1.0f : 0.0f;
        out.w = ((fkey(sc.w) >= t1) || (fkey(gumbelf(row, i0 + 3, 1u)) >= t2)) ? 1.0f : 0.0f;
        *(float4*)(mask + (size_t)row * NACT + i0) = out;
    } else {
        for (int i = i0; i < NACT; ++i) {
            float scv = scores[(size_t)row * NACT + i];
            mask[(size_t)row * NACT + i] =
                ((fkey(scv) >= t1) || (fkey(gumbelf(row, i, 1u)) >= t2)) ? 1.0f : 0.0f;
        }
    }
}

// =====================================================================
// Host launcher. ws layout: [B*N f32 scores][2*B u32 thresholds]
// (needs ~410 MB workspace)
// =====================================================================
extern "C" void kernel_launch(void* const* d_in, const int* in_sizes, int n_in,
                              void* d_out, int out_size, void* d_ws, size_t ws_size,
                              hipStream_t stream)
{
    const float* s    = (const float*)d_in[0];   // [1024, 64]
    const float* embd = (const float*)d_in[1];   // [100000, 64]
    float* mask   = (float*)d_out;               // [1024, 100000]
    float* scores = (float*)d_ws;
    uint32_t* thr = (uint32_t*)((char*)d_ws + (size_t)B_ROWS * NACT * sizeof(float));

    dim3 g1((NACT + TILE_N - 1) / TILE_N, B_ROWS / 16);       // 1563 x 64
    score_gemm_kernel<<<g1, 128, 0, stream>>>(s, embd, scores);

    dim3 g2(B_ROWS, 2);                                       // 1024 rows x 2 streams
    select_threshold_kernel<<<g2, 256, 0, stream>>>(scores, thr);

    dim3 g3((NACT + 1023) / 1024, B_ROWS);                    // 98 x 1024
    mask_kernel<<<g3, 256, 0, stream>>>(scores, thr, mask);
}
